// AutocorrelationCorrelogram_13340168421715
// MI455X (gfx1250) — matrix operand order repaired and checked by reading, not compiled, not run on hardware
//
#include <hip/hip_runtime.h>
#include <hip/hip_bf16.h>

// AutocorrelationCorrelogram on gfx1250 via DFT-as-GEMM on WMMA f16 units.
//
//   frames (300 x 512, Hann)  --Y·C / Y·S-->  Re,Im (f32 accum)
//   P = (Re^2+Im^2)/512 (f16, LDS)  --P·W-->  ACF (f32)
//   relu, 1/sqrt(acf0) normalize, mean over 2 channels, store f32.
//
// Spectrum padded 257 -> 320 bins (zero cols/rows) so 8 waves split the
// column tiles evenly with compile-time trip counts (no spilled acc arrays).
// Workspace: Bf = [2*320][512] f16 (cos|sin, transposed, K-major)  655360 B
//            Wi = [256][320]  f16 (irFFT weights, transposed)      163840 B

typedef __attribute__((ext_vector_type(16))) _Float16 v16h;
typedef __attribute__((ext_vector_type(8)))  _Float16 v8h;
typedef __attribute__((ext_vector_type(8)))  float    v8f;

#define LF    512            // frame length
#define NF    300            // frames per signal
#define NLAG  256            // output lags
#define NKP   320            // padded spectrum bins (20 col tiles of 16)
#define AS    520            // A-tile LDS stride in halves (pad vs banks)
#define PS    NKP            // P-tile LDS stride in halves

__global__ void fill_tables_k(_Float16* __restrict__ Bf, _Float16* __restrict__ Wi)
{
    const double TWO_PI = 6.283185307179586476925286766559;
    int idx = blockIdx.x * blockDim.x + threadIdx.x;
    if (idx < 2 * NKP * LF) {
        // Bf: rows 0..319 = cos columns (transposed), rows 320..639 = sin
        int c = idx >> 9;            // /512
        int n = idx & (LF - 1);
        bool isSin = (c >= NKP);
        int k = isSin ? (c - NKP) : c;
        double v = 0.0;
        if (k <= LF / 2) {
            int ph = (k * n) & (LF - 1);          // exact phase mod 512
            double ang = TWO_PI * (double)ph / (double)LF;
            v = isSin ? sin(ang) : cos(ang);
        }
        Bf[idx] = (_Float16)v;
    } else {
        int j = idx - 2 * NKP * LF;
        if (j < NLAG * NKP) {
            // Wi[lag][k] = w_k * cos(2*pi*k*lag/512); P already carries 1/512
            int lag = j / NKP;
            int k   = j - lag * NKP;
            double v = 0.0;
            if (k <= LF / 2) {
                double wk = (k == 0 || k == LF / 2) ? 1.0 : 2.0;
                int ph = (k * lag) & (LF - 1);
                v = wk * cos(TWO_PI * (double)ph / (double)LF);
            }
            Wi[j] = (_Float16)v;
        }
    }
}

__device__ __forceinline__ v16h load_a_frag(const _Float16* rowbase, int baseK)
{
    // A 16x32 fragment in the k order this code uses for A and B: lanes 0-15 K 0..15, lanes 16-31 K 16..31 (contiguous)
    v8h x = *(const v8h*)(rowbase + baseK);
    v8h y = *(const v8h*)(rowbase + baseK + 8);
    return __builtin_shufflevector(x, y, 0,1,2,3,4,5,6,7,8,9,10,11,12,13,14,15);
}

__device__ __forceinline__ v8f wmma_f16(v16h a, v16h b, v8f c)
{
    return __builtin_amdgcn_wmma_f32_16x16x32_f16(
        /*neg_a=*/false, a, /*neg_b=*/false, b,
        /*c_mod=*/(short)0, c, /*reuse_a=*/false, /*reuse_b=*/false);
}

__global__ __launch_bounds__(256)
void correlogram_k(const float* __restrict__ in,
                   const _Float16* __restrict__ Bf,
                   const _Float16* __restrict__ Wi,
                   float* __restrict__ out)
{
    __shared__ __align__(32) _Float16 Atile[32 * AS];   // 16 frames x 2ch x 512
    __shared__ __align__(32) _Float16 Ptile[32 * PS];   // power spectrum tile
    __shared__ float normLds[32];

    const int a    = blockIdx.y;      // 0..199 = (i*50+j)
    const int tile = blockIdx.x;      // 0..18, frames tile*16 .. tile*16+15
    const int tid  = threadIdx.x;

    // ---- Stage 1: gather + Hann window -> f16 LDS tile (row = 2*frame+ch) ----
    {
        int r    = tid >> 3;          // 0..31
        int part = tid & 7;
        int fl   = r >> 1;
        int ch   = r & 1;
        int m    = tile * 16 + fl; if (m > NF - 1) m = NF - 1;
        long start = (long)((19488.0 * (double)m) / 299.0);   // np.linspace.astype(int64)
        const float* src = in + (((size_t)a) * 20000 + (size_t)start) * 2 + ch;
        int n0 = part * 64;
        #pragma unroll 8
        for (int n = n0; n < n0 + 64; ++n) {
            float w = 0.5f - 0.5f * __cosf(6.28318530717958647f * (float)n * (1.0f / 512.0f));
            Atile[r * AS + n] = (_Float16)(src[2 * n] * w);
        }
    }
    __syncthreads();

    const int lane = tid & 31;
    const int wave = tid >> 5;
    const int rt   = wave & 1;        // row tile: rows rt*16 .. rt*16+15
    const int cg   = wave >> 1;       // column group 0..3
    const int lcol = lane & 15;
    const int lhi  = lane >> 4;

    // ---- Stage 2: forward DFT GEMM (Re and Im accumulators per col tile) ----
    v8f cacc[5] = {}; v8f sacc[5] = {};
    const _Float16* arow = &Atile[(rt * 16 + lcol) * AS];
    #pragma unroll 1
    for (int ks = 0; ks < LF / 32; ++ks) {
        v16h af = load_a_frag(arow, ks * 32 + lhi * 16);
        #pragma unroll
        for (int i = 0; i < 5; ++i) {
            int c = cg + 4 * i;       // col tile 0..19, 5 per group
            const _Float16* bc = Bf + ((size_t)(        c * 16 + lcol)) * LF + ks * 32 + lhi * 16;
            const _Float16* bs = Bf + ((size_t)(NKP   + c * 16 + lcol)) * LF + ks * 32 + lhi * 16;
            cacc[i] = wmma_f16(af, *(const v16h*)bc, cacc[i]);
            sacc[i] = wmma_f16(af, *(const v16h*)bs, sacc[i]);
        }
    }

    // ---- power spectrum / 512 -> f16 LDS (padded cols are exactly zero) ----
    #pragma unroll
    for (int i = 0; i < 5; ++i) {
        int col = (cg + 4 * i) * 16 + lcol;
        #pragma unroll
        for (int v = 0; v < 8; ++v) {
            int row = rt * 16 + lhi * 8 + v;
            float p = (cacc[i][v] * cacc[i][v] + sacc[i][v] * sacc[i][v]) * (1.0f / 512.0f);
            Ptile[row * PS + col] = (_Float16)p;
        }
    }
    __syncthreads();

    // ---- Stage 3: inverse GEMM  ACF = P x W  (K=320, 10 steps) ----
    v8f acc2[4] = {};
    const _Float16* prow = &Ptile[(rt * 16 + lcol) * PS];
    #pragma unroll 1
    for (int ks = 0; ks < NKP / 32; ++ks) {
        v16h af = load_a_frag(prow, ks * 32 + lhi * 16);
        #pragma unroll
        for (int j = 0; j < 4; ++j) {
            const _Float16* bw = Wi + ((size_t)((cg + 4 * j) * 16 + lcol)) * NKP + ks * 32 + lhi * 16;
            acc2[j] = wmma_f16(af, *(const v16h*)bw, acc2[j]);
        }
    }

    // ---- Stage 4: per-row normalization scale from lag 0 (waves owning col 0) ----
    if (cg == 0 && lcol == 0) {
        #pragma unroll
        for (int v = 0; v < 8; ++v) {
            int row = rt * 16 + lhi * 8 + v;
            float a0 = fmaxf(acc2[0][v], 0.0f);
            normLds[row] = (a0 == 0.0f) ? 1.0f : (1.0f / sqrtf(a0));
        }
    }
    __syncthreads();

    float sc[8];
    #pragma unroll
    for (int v = 0; v < 8; ++v) sc[v] = normLds[rt * 16 + lhi * 8 + v];

    // ---- Stage 5: relu, scale, channel-mean (adjacent VGPR pairs), store ----
    #pragma unroll
    for (int j = 0; j < 4; ++j) {
        int col = (cg + 4 * j) * 16 + lcol;
        #pragma unroll
        for (int k = 0; k < 4; ++k) {
            float v0 = fmaxf(acc2[j][2 * k + 0], 0.0f) * sc[2 * k + 0];
            float v1 = fmaxf(acc2[j][2 * k + 1], 0.0f) * sc[2 * k + 1];
            int fl = rt * 8 + lhi * 4 + k;        // local frame index 0..15
            int m  = tile * 16 + fl;
            if (m < NF)
                out[(((size_t)a) * NF + m) * NLAG + col] = 0.5f * (v0 + v1);
        }
    }
}

extern "C" void kernel_launch(void* const* d_in, const int* in_sizes, int n_in,
                              void* d_out, int out_size, void* d_ws, size_t ws_size,
                              hipStream_t stream) {
    const float* in  = (const float*)d_in[0];       // (4,50,20000,2) f32
    float*       out = (float*)d_out;               // (4,50,300,256) f32

    _Float16* Bf = (_Float16*)d_ws;                 // [640][512] halves
    _Float16* Wi = Bf + 2 * NKP * LF;               // [256][320] halves
    (void)in_sizes; (void)n_in; (void)out_size; (void)ws_size;

    int tot = 2 * NKP * LF + NLAG * NKP;            // 409600 table elems
    fill_tables_k<<<(tot + 255) / 256, 256, 0, stream>>>(Bf, Wi);

    dim3 grid((NF + 15) / 16, 200);                 // 19 frame tiles x 200 signals
    correlogram_k<<<grid, 256, 0, stream>>>(in, Bf, Wi, out);
}
